// WindowedAttention_86887188398431
// MI455X (gfx1250) — compile-verified, matrix operand order checked by reading, not run on hardware
//
#include <hip/hip_runtime.h>
#include <hip/hip_bf16.h>

// Problem constants (B=1)
#define T_SEQ   4096
#define D_MODEL 1024
#define N_HEADS 8
#define HEAD_DIM 64
#define INNER   512           // H*HD
#define QKV_N   1536          // 3*INNER
#define WIN     256

typedef __attribute__((ext_vector_type(16))) __bf16 v16bf;
typedef __attribute__((ext_vector_type(8)))  float  v8f;

union FragU { uint4 u[2]; v16bf v; };

// float -> bf16 (round-to-nearest-even), pure bit manipulation (storage-only __bf16)
__device__ inline __bf16 f2bf(float f) {
    union { float f; unsigned u; } a; a.f = f;
    unsigned r = a.u + 0x7FFFu + ((a.u >> 16) & 1u);
    union { unsigned short s; __bf16 b; } o; o.s = (unsigned short)(r >> 16);
    return o.b;
}

// Async global->LDS 16B copy (GLOBAL_LOAD_ASYNC_TO_LDS_B128, ASYNCcnt-tracked).
__device__ inline void async_b128(const __bf16* g, __bf16* lds) {
    unsigned ldsoff = (unsigned)(size_t)(void*)lds;
    asm volatile("global_load_async_to_lds_b128 %0, %1, off"
                 :: "v"(ldsoff), "v"(g) : "memory");
}
__device__ inline void wait_async0() {
    asm volatile("s_wait_asynccnt 0x0" ::: "memory");
}

// Load one 16x32 bf16 A/B fragment (ISA 7.12.2 layout): two contiguous 16B chunks per lane.
__device__ inline v16bf load_frag(const __bf16* base, int row, int stride, int kbase) {
    FragU f;
    f.u[0] = *reinterpret_cast<const uint4*>(base + (size_t)row * stride + kbase);
    f.u[1] = *reinterpret_cast<const uint4*>(base + (size_t)row * stride + kbase + 16);
    return f.v;
}

__device__ inline v8f wmma_bf16(v16bf a, v16bf b, v8f c) {
    return __builtin_amdgcn_wmma_f32_16x16x32_bf16(false, a, false, b, (short)0, c, false, false);
}

__device__ inline v8f zero8() {
    v8f z;
    #pragma unroll
    for (int i = 0; i < 8; ++i) z[i] = 0.f;
    return z;
}

// ---------------------------------------------------------------- convert f32 -> bf16 (plain)
__global__ void cvt_f32_bf16(const float* __restrict__ in, __bf16* __restrict__ out, int n) {
    int i = blockIdx.x * blockDim.x + threadIdx.x;
    if (i < n) out[i] = f2bf(in[i]);
}

// ---------------------------------------------------------------- convert + transpose f32[K][N] -> bf16[N][K]
__global__ __launch_bounds__(256) void cvt_transpose_f32_bf16(
        const float* __restrict__ in, __bf16* __restrict__ out, int K, int N) {
    __shared__ float tile[32][33];
    const int k0 = blockIdx.y * 32, n0 = blockIdx.x * 32;
    const int tx = threadIdx.x & 31, ty = threadIdx.x >> 5;
    #pragma unroll
    for (int i = 0; i < 32; i += 8)
        tile[ty + i][tx] = in[(size_t)(k0 + ty + i) * N + n0 + tx];
    __syncthreads();
    #pragma unroll
    for (int i = 0; i < 32; i += 8)
        out[(size_t)(n0 + ty + i) * K + k0 + tx] = f2bf(tile[tx][ty + i]);
}

// ---------------------------------------------------------------- bf16 GEMM, C f32
// C[M,N] = A[M,K] * Bt[N,K]^T  (B pre-transposed).  Block tile 128x128,
// 256 threads = 8 waves (2x4), wave tile 64x32, K step 32.
// Double-buffered LDS; async copies for step i+1 overlap WMMA compute of step i.
__global__ __launch_bounds__(256) void gemm_bf16(
        const __bf16* __restrict__ A, const __bf16* __restrict__ Bt,
        float* __restrict__ C, int M, int N, int K) {
    __shared__ __align__(16) __bf16 Atile[2][128][32];
    __shared__ __align__(16) __bf16 Btile[2][128][32];
    const int t  = threadIdx.x;
    const int w  = t >> 5, L = t & 31;
    const int hi = L >> 4, ln = L & 15, kb = hi * 8;
    const int wm = w >> 2, wn = w & 3;
    const int m0 = blockIdx.y * 128, n0 = blockIdx.x * 128;

    v8f acc[4][2];
    #pragma unroll
    for (int a = 0; a < 4; ++a)
        #pragma unroll
        for (int b = 0; b < 2; ++b) acc[a][b] = zero8();

    auto stage = [&](int k0, int buf) {
        #pragma unroll
        for (int i = 0; i < 2; ++i) {
            int idx = t + i * 256;                 // 0..511 (128 rows x 4 chunks)
            int row = idx >> 2, part = idx & 3;
            async_b128(A  + (size_t)(m0 + row) * K + k0 + part * 8, &Atile[buf][row][part * 8]);
            async_b128(Bt + (size_t)(n0 + row) * K + k0 + part * 8, &Btile[buf][row][part * 8]);
        }
    };

    const int nk = K / 32;
    stage(0, 0);
    for (int ik = 0; ik < nk; ++ik) {
        const int buf = ik & 1;
        wait_async0();                 // my slice of batch ik has landed
        __syncthreads();               // everyone's batch ik visible; buf^1 free to overwrite
        if (ik + 1 < nk) stage((ik + 1) * 32, buf ^ 1);   // overlaps with compute below

        v16bf bfr[2];
        #pragma unroll
        for (int nt = 0; nt < 2; ++nt)
            bfr[nt] = load_frag(&Btile[buf][0][0], wn * 32 + nt * 16 + ln, 32, kb);
        #pragma unroll
        for (int mt = 0; mt < 4; ++mt) {
            v16bf af = load_frag(&Atile[buf][0][0], wm * 64 + mt * 16 + ln, 32, kb);
            #pragma unroll
            for (int nt = 0; nt < 2; ++nt)
                acc[mt][nt] = wmma_bf16(af, bfr[nt], acc[mt][nt]);
        }
    }
    #pragma unroll
    for (int mt = 0; mt < 4; ++mt)
        #pragma unroll
        for (int nt = 0; nt < 2; ++nt)
            #pragma unroll
            for (int r = 0; r < 8; ++r) {
                int row = m0 + wm * 64 + mt * 16 + r + 8 * hi;
                int col = n0 + wn * 32 + nt * 16 + ln;
                C[(size_t)row * N + col] = acc[mt][nt][r];
            }
}

// ---------------------------------------------------------------- RoPE + split
// Q,K -> bf16 [H][T][64] (row-major).  V -> bf16 [H][64][T] (dim-major).
__global__ void rope_split(const float* __restrict__ qkv,
                           __bf16* __restrict__ Qb, __bf16* __restrict__ Kb,
                           __bf16* __restrict__ Vt) {
    int gid = blockIdx.x * blockDim.x + threadIdx.x;   // T*H*32 threads
    int d = gid & 31;
    int h = (gid >> 5) & 7;
    int tt = gid >> 8;
    if (tt >= T_SEQ) return;
    const float* base = qkv + (size_t)tt * QKV_N;
    float inv = __expf(-((float)(2 * d) / 64.f) * 9.210340371976184f);  // 10000^(-2d/64)
    float ang = (float)tt * inv;
    float s, c; __sincosf(ang, &s, &c);
    int off = h * 64 + d;
    float q1 = base[off],        q2 = base[off + 32];
    float k1 = base[512 + off],  k2 = base[512 + off + 32];
    float v1 = base[1024 + off], v2 = base[1024 + off + 32];
    size_t o = ((size_t)h * T_SEQ + tt) * 64 + d;
    Qb[o]      = f2bf(q1 * c - q2 * s);
    Qb[o + 32] = f2bf(q1 * s + q2 * c);
    Kb[o]      = f2bf(k1 * c - k2 * s);
    Kb[o + 32] = f2bf(k1 * s + k2 * c);
    Vt[((size_t)h * 64 + d) * T_SEQ + tt]        = f2bf(v1);
    Vt[((size_t)h * 64 + d + 32) * T_SEQ + tt]   = f2bf(v2);
}

// ---------------------------------------------------------------- windowed flash attention
// Workgroup = (head, 64 queries) = 4 waves x 16 queries. 32-key chunks streamed via
// double-buffered LDS with one-chunk-ahead async copies.
__global__ __launch_bounds__(128) void attn_win(
        const __bf16* __restrict__ Qb, const __bf16* __restrict__ Kb,
        const __bf16* __restrict__ Vg, __bf16* __restrict__ Ob) {
    __shared__ __align__(16) __bf16 Kc[2][32][64];     // key chunk [key][dim]
    __shared__ __align__(16) __bf16 Vc[2][64][32];     // value chunk [dim][key]
    __shared__ __align__(16) __bf16 Pscr[4][16][32];   // per-wave P transpose scratch
    const int h  = blockIdx.y;
    const int q0 = blockIdx.x * 64;
    const int t  = threadIdx.x, w = t >> 5, L = t & 31;
    const int hi = L >> 4, ln = L & 15, kb = hi * 8;
    const int qa = q0 + w * 16;
    const __bf16* Qh = Qb + (size_t)h * T_SEQ * 64;
    const __bf16* Kh = Kb + (size_t)h * T_SEQ * 64;
    const __bf16* Vh = Vg + (size_t)h * 64 * T_SEQ;    // [64][T]
    const float scale = 0.125f;                        // 1/sqrt(64)

    v16bf qf[2];
    #pragma unroll
    for (int c = 0; c < 2; ++c)
        qf[c] = load_frag(Qh, qa + ln, 64, c * 32 + kb);

    v8f o[4]; float m_run[8], l_run[8];
    #pragma unroll
    for (int d = 0; d < 4; ++d) o[d] = zero8();
    #pragma unroll
    for (int r = 0; r < 8; ++r) { m_run[r] = -1e30f; l_run[r] = 0.f; }

    int cstart = q0 - (WIN - 1);
    cstart = (cstart < 0) ? 0 : (cstart & ~31);

    auto stage = [&](int jc, int buf) {
        #pragma unroll
        for (int i = 0; i < 2; ++i) {
            int idx = t + i * 128;
            int krow = idx >> 3, kpart = idx & 7;          // 32 rows x 8 chunks
            async_b128(Kh + (size_t)(jc + krow) * 64 + kpart * 8, &Kc[buf][krow][kpart * 8]);
            int vrow = idx >> 2, vpart = idx & 3;          // 64 rows x 4 chunks
            async_b128(Vh + (size_t)vrow * T_SEQ + jc + vpart * 8, &Vc[buf][vrow][vpart * 8]);
        }
    };

    const int nchunks = (q0 + 64 - cstart) / 32;
    stage(cstart, 0);
    for (int ic = 0; ic < nchunks; ++ic) {
        const int jc = cstart + ic * 32;
        const int buf = ic & 1;
        wait_async0();
        __syncthreads();
        if (ic + 1 < nchunks) stage(jc + 32, buf ^ 1);   // overlaps with compute below

        // S(16q x 32k) = Q K^T
        v8f s0 = zero8(), s1 = zero8();
        #pragma unroll
        for (int c = 0; c < 2; ++c) {
            v16bf kf0 = load_frag(&Kc[buf][0][0], ln,      64, c * 32 + kb);
            v16bf kf1 = load_frag(&Kc[buf][0][0], 16 + ln, 64, c * 32 + kb);
            s0 = wmma_bf16(qf[c], kf0, s0);
            s1 = wmma_bf16(qf[c], kf1, s1);
        }

        // mask + online softmax (rows live in 16-lane halves -> shfl_xor 1/2/4/8)
        int j0 = jc + ln, j1 = j0 + 16;
        #pragma unroll
        for (int r = 0; r < 8; ++r) {
            int qi = qa + r + 8 * hi;
            bool v0 = (j0 <= qi) && (j0 > qi - WIN);
            bool v1 = (j1 <= qi) && (j1 > qi - WIN);
            float a0 = v0 ? s0[r] * scale : -1e30f;
            float a1 = v1 ? s1[r] * scale : -1e30f;
            float rm = fmaxf(a0, a1);
            rm = fmaxf(rm, __shfl_xor(rm, 1));
            rm = fmaxf(rm, __shfl_xor(rm, 2));
            rm = fmaxf(rm, __shfl_xor(rm, 4));
            rm = fmaxf(rm, __shfl_xor(rm, 8));
            float mnew  = fmaxf(m_run[r], rm);
            float alpha = __expf(m_run[r] - mnew);
            float p0 = v0 ? __expf(a0 - mnew) : 0.f;
            float p1 = v1 ? __expf(a1 - mnew) : 0.f;
            float rs = p0 + p1;
            rs += __shfl_xor(rs, 1);
            rs += __shfl_xor(rs, 2);
            rs += __shfl_xor(rs, 4);
            rs += __shfl_xor(rs, 8);
            l_run[r] = l_run[r] * alpha + rs;
            m_run[r] = mnew;
            #pragma unroll
            for (int d = 0; d < 4; ++d) o[d][r] = o[d][r] * alpha;
            int m = r + 8 * hi;
            Pscr[w][m][ln]      = f2bf(p0);
            Pscr[w][m][ln + 16] = f2bf(p1);
        }
        // same-wave LDS is in-order; just stop compiler reordering the punned reload
        asm volatile("" ::: "memory");

        // O += P @ V
        v16bf pf = load_frag(&Pscr[w][0][0], ln, 32, kb);
        #pragma unroll
        for (int d = 0; d < 4; ++d) {
            v16bf vf = load_frag(&Vc[buf][0][0], d * 16 + ln, 32, kb);
            o[d] = wmma_bf16(pf, vf, o[d]);
        }
    }

    // normalize and store bf16 into attn_out [T][INNER], col = h*64 + dim
    #pragma unroll
    for (int r = 0; r < 8; ++r) {
        float invl = 1.f / (l_run[r] + 1e-20f);
        int row = qa + r + 8 * hi;
        #pragma unroll
        for (int d = 0; d < 4; ++d) {
            int col = h * 64 + d * 16 + ln;
            Ob[(size_t)row * INNER + col] = f2bf(o[d][r] * invl);
        }
    }
}

// ---------------------------------------------------------------- launch
extern "C" void kernel_launch(void* const* d_in, const int* in_sizes, int n_in,
                              void* d_out, int out_size, void* d_ws, size_t ws_size,
                              hipStream_t stream) {
    const float* x    = (const float*)d_in[0];
    const float* Wqkv = (const float*)d_in[1];
    const float* Wout = (const float*)d_in[2];
    float* out = (float*)d_out;

    char* ws = (char*)d_ws;
    size_t off = 0;
    auto carve = [&](size_t bytes) -> void* {
        void* p = ws + off;
        off += (bytes + 255) & ~(size_t)255;
        return p;
    };
    __bf16* xb     = (__bf16*)carve((size_t)T_SEQ * D_MODEL * 2);
    __bf16* wqkvT  = (__bf16*)carve((size_t)D_MODEL * QKV_N * 2);   // [QKV_N][D_MODEL]
    __bf16* woutT  = (__bf16*)carve((size_t)INNER * D_MODEL * 2);   // [D_MODEL][INNER]
    float*  qkvf   = (float*) carve((size_t)T_SEQ * QKV_N * 4);
    __bf16* Qb     = (__bf16*)carve((size_t)N_HEADS * T_SEQ * HEAD_DIM * 2);
    __bf16* Kb     = (__bf16*)carve((size_t)N_HEADS * T_SEQ * HEAD_DIM * 2);
    __bf16* Vt     = (__bf16*)carve((size_t)N_HEADS * HEAD_DIM * T_SEQ * 2);
    __bf16* attnb  = (__bf16*)carve((size_t)T_SEQ * INNER * 2);
    (void)ws_size; (void)n_in; (void)in_sizes; (void)out_size;

    int n1 = T_SEQ * D_MODEL;
    cvt_f32_bf16<<<(n1 + 255) / 256, 256, 0, stream>>>(x, xb, n1);
    cvt_transpose_f32_bf16<<<dim3(QKV_N / 32, D_MODEL / 32), 256, 0, stream>>>(
        Wqkv, wqkvT, D_MODEL, QKV_N);
    cvt_transpose_f32_bf16<<<dim3(D_MODEL / 32, INNER / 32), 256, 0, stream>>>(
        Wout, woutT, INNER, D_MODEL);

    gemm_bf16<<<dim3(QKV_N / 128, T_SEQ / 128), 256, 0, stream>>>(
        xb, wqkvT, qkvf, T_SEQ, QKV_N, D_MODEL);

    int nrope = T_SEQ * N_HEADS * 32;
    rope_split<<<(nrope + 255) / 256, 256, 0, stream>>>(qkvf, Qb, Kb, Vt);

    attn_win<<<dim3(T_SEQ / 64, N_HEADS), 128, 0, stream>>>(Qb, Kb, Vt, attnb);

    gemm_bf16<<<dim3(D_MODEL / 128, T_SEQ / 128), 256, 0, stream>>>(
        attnb, woutT, out, T_SEQ, D_MODEL, INNER);
}
